// Attention_29738353557849
// MI455X (gfx1250) — compile-verified, matrix operand order checked by reading, not run on hardware
//
#include <hip/hip_runtime.h>

// ---------------------------------------------------------------------------
// CDNA5 (gfx1250) wave32 WMMA attention block.
//   K0a/b/c: one-time f32 -> bf16 conversion of x, qkv_w, proj_w
//   K0d:     bias matrix gather  bmat[h,n,m] = biases[h, idxs[n,m]]
//   K1: qkv = BN(x @ qkv_w^T)  -> Q(bf16, *SCALE), K(bf16), V^T(bf16)
//   K2: flash attention per (b,h) + SiLU -> sbuf bf16 [B,N,2048]
//   K3: out = BN(sbuf @ proj_w^T)  (f32)
// Inner GEMM loops are written load-all / compute-all so operands live in
// disjoint VGPRs and the scheduler can overlap global_load_b128 with WMMA
// via partial s_wait_loadcnt instead of wait-zero serialization.
// ---------------------------------------------------------------------------

typedef __attribute__((ext_vector_type(16))) __bf16 v16bf;
typedef __attribute__((ext_vector_type(8)))  float  v8f;

union BfOp {
  v16bf v;
  unsigned short h[16];
  uint4 q[2];
};

__device__ __forceinline__ unsigned short f2bf(float f) {
  unsigned int u = __float_as_uint(f);
  u += 0x7fffu + ((u >> 16) & 1u);          // round-to-nearest-even
  return (unsigned short)(u >> 16);
}

__device__ __forceinline__ v8f wmma_bf16(const BfOp& a, const BfOp& b, v8f c) {
  return __builtin_amdgcn_wmma_f32_16x16x32_bf16(
      /*neg_a=*/false, a.v, /*neg_b=*/false, b.v,
      /*c_mod=*/(short)0, c, /*reuse_a=*/false, /*reuse_b=*/false);
}

// ---------------------------------------------------------------------------
// Kernel 0: bulk f32 -> bf16 (8 elements / thread, b128 in / b128 out)
// ---------------------------------------------------------------------------
__global__ __launch_bounds__(256)
void cvt_bf16_kernel(const float* __restrict__ in, unsigned short* __restrict__ out,
                     int n8) {
  const int i = blockIdx.x * 256 + threadIdx.x;
  if (i >= n8) return;
  const float4 a = *(const float4*)(in + (size_t)i * 8);
  const float4 b = *(const float4*)(in + (size_t)i * 8 + 4);
  union { unsigned short h[8]; uint4 q; } u;
  u.h[0] = f2bf(a.x); u.h[1] = f2bf(a.y); u.h[2] = f2bf(a.z); u.h[3] = f2bf(a.w);
  u.h[4] = f2bf(b.x); u.h[5] = f2bf(b.y); u.h[6] = f2bf(b.z); u.h[7] = f2bf(b.w);
  *(uint4*)(out + (size_t)i * 8) = u.q;
}

// ---------------------------------------------------------------------------
// Kernel 0d: positional-bias gather, done once (idxs is shared by all (b,h)).
// ---------------------------------------------------------------------------
__global__ __launch_bounds__(256)
void bias_prep_kernel(const float* __restrict__ biases, const int* __restrict__ idxs,
                      float* __restrict__ bmat) {
  const int i = blockIdx.x * 256 + threadIdx.x;   // over 8*784*784 (exact grid)
  const int h = i / (784 * 784);
  const int nm = i - h * (784 * 784);
  bmat[i] = biases[h * 784 + idxs[nm]];
}

// ---------------------------------------------------------------------------
// Kernel 1: QKV GEMM (bf16 operands) + BN, scatter into Q / K / V^T.
// grid (7, 24, 16), block 256 (8 waves). Wave = 16 rows x 128 cols.
// ---------------------------------------------------------------------------
__global__ __launch_bounds__(256)
void qkv_bn_kernel(const unsigned short* __restrict__ xb,
                   const unsigned short* __restrict__ wb,
                   const float* __restrict__ gam, const float* __restrict__ bet,
                   const float* __restrict__ mu,  const float* __restrict__ var,
                   unsigned short* __restrict__ qb, unsigned short* __restrict__ kb,
                   unsigned short* __restrict__ vt) {
  const int lane = threadIdx.x & 31;
  const int wave = threadIdx.x >> 5;
  const int lo = lane & 15;
  const int hx = lane >> 4;                 // half-wave
  const int b = blockIdx.z;
  const int m0 = blockIdx.x * 128 + wave * 16;   // token-row tile
  if (m0 >= 784) return;                    // tiles all-or-nothing (784 = 49*16)
  const int o0 = blockIdx.y * 128;          // output-column tile

  v8f acc[8];
#pragma unroll
  for (int j = 0; j < 8; ++j) acc[j] = (v8f)(0.0f);

  const unsigned short* xrow = xb + ((size_t)b * 784 + (m0 + lo)) * 512;
  for (int kk = 0; kk < 512; kk += 32) {
    BfOp a;
    a.q[0] = *(const uint4*)(xrow + kk + 8 * hx);
    a.q[1] = *(const uint4*)(xrow + kk + 16 + 8 * hx);
    BfOp bops[8];
#pragma unroll
    for (int j = 0; j < 8; ++j) {
      const unsigned short* wrow = wb + (size_t)(o0 + j * 16 + lo) * 512 + kk;
      bops[j].q[0] = *(const uint4*)(wrow + 8 * hx);
      bops[j].q[1] = *(const uint4*)(wrow + 16 + 8 * hx);
    }
#pragma unroll
    for (int j = 0; j < 8; ++j) acc[j] = wmma_bf16(a, bops[j], acc[j]);
  }

#pragma unroll
  for (int j = 0; j < 8; ++j) {
    const int o = o0 + j * 16 + lo;
    const float s  = gam[o] * rsqrtf(var[o] + 1e-5f);
    const float mn = mu[o], bt = bet[o];
    const int hh = o / 384;
    const int jj = o % 384;
#pragma unroll
    for (int r = 0; r < 8; ++r) {
      const int n = m0 + r + 8 * hx;
      const float v = (acc[j][r] - mn) * s + bt;
      if (jj < 64) {
        qb[(((size_t)b * 8 + hh) * 784 + n) * 64 + jj] = f2bf(v * 0.125f); // fold SCALE
      } else if (jj < 128) {
        kb[(((size_t)b * 8 + hh) * 784 + n) * 64 + (jj - 64)] = f2bf(v);
      } else {
        vt[(((size_t)b * 8 + hh) * 256 + (jj - 128)) * 784 + n] = f2bf(v); // transposed
      }
    }
  }
}

// ---------------------------------------------------------------------------
// Kernel 2: flash attention per (b,h), 16 q-rows per wave, 32 keys/step.
// grid (13, 128), block 128 (4 waves, each an independent q-tile).
// ---------------------------------------------------------------------------
__global__ __launch_bounds__(128)
void attn_kernel(const unsigned short* __restrict__ qb,
                 const unsigned short* __restrict__ kb,
                 const unsigned short* __restrict__ vt,
                 const float* __restrict__ bmat,   // [8, 784, 784] f32
                 unsigned short* __restrict__ sb) {
  __shared__ __align__(16) unsigned short pshare[4 * 16 * 32];
  const int lane = threadIdx.x & 31;
  const int wave = threadIdx.x >> 5;
  const int lo = lane & 15;
  const int hx = lane >> 4;
  const int bh = blockIdx.y;                // b*8 + h
  const int h = bh & 7;
  int qt = blockIdx.x * 4 + wave;
  const bool valid = (qt < 49);
  if (qt > 48) qt = 48;                     // clamp, keep uniform control flow
  const int nbase = qt * 16;

  // hoist Q A-operands (K = 0..31 and 32..63)
  const unsigned short* qrow = qb + ((size_t)bh * 784 + (nbase + lo)) * 64;
  BfOp qa[2];
#pragma unroll
  for (int t = 0; t < 2; ++t) {
    qa[t].q[0] = *(const uint4*)(qrow + t * 32 + 8 * hx);
    qa[t].q[1] = *(const uint4*)(qrow + t * 32 + 16 + 8 * hx);
  }

  float mrow[8], lrow[8];
#pragma unroll
  for (int r = 0; r < 8; ++r) { mrow[r] = -3.0e38f; lrow[r] = 0.0f; }
  v8f oacc[16];
#pragma unroll
  for (int d = 0; d < 16; ++d) oacc[d] = (v8f)(0.0f);

  unsigned short* pw = pshare + wave * 512; // per-wave P tile (16x32 bf16)
  const float* bias_h = bmat + (size_t)h * 784 * 784;

  for (int m0 = 0; m0 < 784; m0 += 32) {
    // ---- scores: q @ k^T (pre-scaled); batch all 4 K operands ----
    BfOp kops[4];
#pragma unroll
    for (int j = 0; j < 2; ++j) {
      const int m = m0 + j * 16 + lo;
      const int mc = m < 784 ? m : 783;
      const unsigned short* krow = kb + ((size_t)bh * 784 + mc) * 64;
      kops[2 * j + 0].q[0] = *(const uint4*)(krow + 8 * hx);
      kops[2 * j + 0].q[1] = *(const uint4*)(krow + 16 + 8 * hx);
      kops[2 * j + 1].q[0] = *(const uint4*)(krow + 32 + 8 * hx);
      kops[2 * j + 1].q[1] = *(const uint4*)(krow + 32 + 16 + 8 * hx);
    }
    v8f sc[2];
#pragma unroll
    for (int j = 0; j < 2; ++j) {
      sc[j] = wmma_bf16(qa[0], kops[2 * j + 0], (v8f)(0.0f));
      sc[j] = wmma_bf16(qa[1], kops[2 * j + 1], sc[j]);
    }
    // ---- + positional bias (precomputed, coalesced); mask OOB keys ----
#pragma unroll
    for (int j = 0; j < 2; ++j) {
      const int m = m0 + j * 16 + lo;
#pragma unroll
      for (int r = 0; r < 8; ++r) {
        const int n = nbase + r + 8 * hx;
        float bv = -3.0e38f;
        if (m < 784) bv = bias_h[n * 784 + m];
        sc[j][r] += bv;
      }
    }
    // ---- online softmax (row stats live in one VGPR per 16-lane half) ----
#pragma unroll
    for (int r = 0; r < 8; ++r) {
      float mx = fmaxf(sc[0][r], sc[1][r]);
#pragma unroll
      for (int off = 8; off >= 1; off >>= 1)
        mx = fmaxf(mx, __shfl_xor(mx, off, 16));
      const float mn = fmaxf(mrow[r], mx);
      const float corr = __expf(mrow[r] - mn);
      mrow[r] = mn;
      lrow[r] *= corr;
#pragma unroll
      for (int d = 0; d < 16; ++d) oacc[d][r] *= corr;
      const float p0 = __expf(sc[0][r] - mn);
      const float p1 = __expf(sc[1][r] - mn);
      sc[0][r] = p0; sc[1][r] = p1;
      float s = p0 + p1;
#pragma unroll
      for (int off = 8; off >= 1; off >>= 1)
        s += __shfl_xor(s, off, 16);
      lrow[r] += s;
    }
    // ---- P: C-layout -> LDS -> A-layout bf16 ----
    __syncthreads();
#pragma unroll
    for (int j = 0; j < 2; ++j)
#pragma unroll
      for (int r = 0; r < 8; ++r)
        pw[(r + 8 * hx) * 32 + j * 16 + lo] = f2bf(sc[j][r]);
    __syncthreads();
    BfOp pa;
    pa.q[0] = *(const uint4*)(pw + lo * 32 + 8 * hx);
    pa.q[1] = *(const uint4*)(pw + lo * 32 + 16 + 8 * hx);
    // ---- oacc += P @ V : groups of 4 batched V operands ----
#pragma unroll
    for (int g = 0; g < 4; ++g) {
      BfOp vb[4];
#pragma unroll
      for (int t = 0; t < 4; ++t) {
        const int d = g * 4 + t;
        const unsigned short* vrow =
            vt + ((size_t)bh * 256 + d * 16 + lo) * 784 + m0;
        vb[t].q[0] = *(const uint4*)(vrow + 8 * hx);
        vb[t].q[1] = *(const uint4*)(vrow + 16 + 8 * hx);
      }
#pragma unroll
      for (int t = 0; t < 4; ++t)
        oacc[g * 4 + t] = wmma_bf16(pa, vb[t], oacc[g * 4 + t]);
    }
  }

  if (valid) {
    const int b = bh >> 3;
    float rinv[8];
#pragma unroll
    for (int r = 0; r < 8; ++r) rinv[r] = 1.0f / lrow[r];
#pragma unroll
    for (int d = 0; d < 16; ++d) {
      const int col = h * 256 + d * 16 + lo;
#pragma unroll
      for (int r = 0; r < 8; ++r) {
        const int n = nbase + r + 8 * hx;
        float o = oacc[d][r] * rinv[r];
        o = o / (1.0f + __expf(-o));        // SiLU
        sb[((size_t)b * 784 + n) * 2048 + col] = f2bf(o);
      }
    }
  }
}

// ---------------------------------------------------------------------------
// Kernel 3: proj GEMM (bf16 operands) + BN.
// grid 784 (16 rows each), block 256 (8 waves x 64 cols).
// ---------------------------------------------------------------------------
__global__ __launch_bounds__(256)
void proj_bn_kernel(const unsigned short* __restrict__ sb,
                    const unsigned short* __restrict__ pwb,
                    const float* __restrict__ gam, const float* __restrict__ bet,
                    const float* __restrict__ mu,  const float* __restrict__ var,
                    float* __restrict__ out) {
  const int lane = threadIdx.x & 31;
  const int wave = threadIdx.x >> 5;
  const int lo = lane & 15;
  const int hx = lane >> 4;
  const int row0 = blockIdx.x * 16;         // of 12544 = 16*784 rows (exact)
  const int c0 = wave * 64;

  v8f acc[4];
#pragma unroll
  for (int j = 0; j < 4; ++j) acc[j] = (v8f)(0.0f);

  const unsigned short* srow = sb + (size_t)(row0 + lo) * 2048;
  for (int kk = 0; kk < 2048; kk += 32) {
    BfOp a;
    a.q[0] = *(const uint4*)(srow + kk + 8 * hx);
    a.q[1] = *(const uint4*)(srow + kk + 16 + 8 * hx);
    BfOp bops[4];
#pragma unroll
    for (int j = 0; j < 4; ++j) {
      const unsigned short* wrow = pwb + (size_t)(c0 + j * 16 + lo) * 2048 + kk;
      bops[j].q[0] = *(const uint4*)(wrow + 8 * hx);
      bops[j].q[1] = *(const uint4*)(wrow + 16 + 8 * hx);
    }
#pragma unroll
    for (int j = 0; j < 4; ++j) acc[j] = wmma_bf16(a, bops[j], acc[j]);
  }

#pragma unroll
  for (int j = 0; j < 4; ++j) {
    const int c = c0 + j * 16 + lo;
    const float s  = gam[c] * rsqrtf(var[c] + 1e-5f);
    const float mn = mu[c], bt = bet[c];
#pragma unroll
    for (int r = 0; r < 8; ++r) {
      const int row = row0 + r + 8 * hx;
      out[(size_t)row * 512 + c] = (acc[j][r] - mn) * s + bt;
    }
  }
}

// ---------------------------------------------------------------------------
extern "C" void kernel_launch(void* const* d_in, const int* in_sizes, int n_in,
                              void* d_out, int out_size, void* d_ws, size_t ws_size,
                              hipStream_t stream) {
  (void)in_sizes; (void)n_in; (void)out_size; (void)ws_size;
  const float* x    = (const float*)d_in[0];
  const float* qw   = (const float*)d_in[1];
  const float* qg   = (const float*)d_in[2];
  const float* qbt  = (const float*)d_in[3];
  const float* qm   = (const float*)d_in[4];
  const float* qv   = (const float*)d_in[5];
  const float* ab   = (const float*)d_in[6];
  const float* pwm  = (const float*)d_in[7];
  const float* pg   = (const float*)d_in[8];
  const float* pb   = (const float*)d_in[9];
  const float* pm   = (const float*)d_in[10];
  const float* pv   = (const float*)d_in[11];
  const int*   idx  = (const int*)d_in[12];

  char* ws = (char*)d_ws;
  const size_t qk_bytes  = (size_t)16 * 8 * 784 * 64 * 2;   // 12.8 MB each (Q, K)
  const size_t vt_bytes  = (size_t)16 * 8 * 256 * 784 * 2;  // 51.4 MB (V^T)
  const size_t sb_bytes  = (size_t)16 * 784 * 2048 * 2;     // 51.4 MB (silu out)
  const size_t xb_bytes  = (size_t)16 * 784 * 512 * 2;      // 12.8 MB
  const size_t wb_bytes  = (size_t)3072 * 512 * 2;          //  3.1 MB
  const size_t pwb_bytes = (size_t)512 * 2048 * 2;          //  2.1 MB

  size_t off = 0;
  unsigned short* qbuf = (unsigned short*)(ws + off); off += qk_bytes;
  unsigned short* kbuf = (unsigned short*)(ws + off); off += qk_bytes;
  unsigned short* vtb  = (unsigned short*)(ws + off); off += vt_bytes;
  unsigned short* sbuf = (unsigned short*)(ws + off); off += sb_bytes;
  unsigned short* xb   = (unsigned short*)(ws + off); off += xb_bytes;
  unsigned short* wb   = (unsigned short*)(ws + off); off += wb_bytes;
  unsigned short* pwb  = (unsigned short*)(ws + off); off += pwb_bytes;
  float*          bmat = (float*)(ws + off);          // 19.7 MB

  // one-time conversions + bias gather
  cvt_bf16_kernel<<<dim3(3136), 256, 0, stream>>>(x,   xb,  802816); // 16*784*512/8
  cvt_bf16_kernel<<<dim3(768),  256, 0, stream>>>(qw,  wb,  196608); // 3072*512/8
  cvt_bf16_kernel<<<dim3(512),  256, 0, stream>>>(pwm, pwb, 131072); // 512*2048/8
  bias_prep_kernel<<<dim3(19208), 256, 0, stream>>>(ab, idx, bmat);  // 8*784*784

  qkv_bn_kernel<<<dim3(7, 24, 16), 256, 0, stream>>>(xb, wb, qg, qbt, qm, qv,
                                                     qbuf, kbuf, vtb);
  attn_kernel<<<dim3(13, 128), 128, 0, stream>>>(qbuf, kbuf, vtb, bmat, sbuf);
  proj_bn_kernel<<<dim3(784), 256, 0, stream>>>(sbuf, pwb, pg, pb, pm, pv,
                                                (float*)d_out);
}
